// LSTMModel_38010460569746
// MI455X (gfx1250) — matrix operand order repaired and checked by reading, not compiled, not run on hardware
//
#include <hip/hip_runtime.h>

#define B_TOT 4096
#define T_LEN 512
#define F_IN  32
#define HID   64
#define HZ    24

typedef __attribute__((ext_vector_type(16))) _Float16 v16h;
typedef __attribute__((ext_vector_type(8)))  float    v8f;

// B-fragment run from fp32: 8 values at p[0..7], 8 at p[16..23]
__device__ __forceinline__ v16h load_cvt_B(const float* __restrict__ p) {
  v16h r;
#pragma unroll
  for (int i = 0; i < 8; ++i) { r[i] = (_Float16)p[i]; r[8 + i] = (_Float16)p[16 + i]; }
  return r;
}

// A-fragment run from fp32: 8 values at p[0..7], 8 at p[16..23]
__device__ __forceinline__ v16h load_cvt_A(const float* __restrict__ p) {
  v16h r;
#pragma unroll
  for (int i = 0; i < 8; ++i) { r[i] = (_Float16)p[i]; r[8 + i] = (_Float16)p[16 + i]; }
  return r;
}

// A-fragment run from LDS f16: 8 halfs at p[0..7], 8 at p[16..23]
__device__ __forceinline__ v16h load_A_lds(const _Float16* p) {
  v16h r;
#pragma unroll
  for (int i = 0; i < 8; ++i) { r[i] = p[i]; r[8 + i] = p[16 + i]; }
  return r;
}

__device__ __forceinline__ float sigm_fast(float x) {
  return 1.0f / (1.0f + __expf(-x));
}
__device__ __forceinline__ float tanh_fast(float x) {
  x = fminf(fmaxf(x, -15.0f), 15.0f);
  float e = __expf(2.0f * x);
  return (e - 1.0f) / (e + 1.0f);
}

__global__ void __launch_bounds__(128, 1)
lstm_fused_wmma(const float* __restrict__ x,    const float* __restrict__ W_ih,
                const float* __restrict__ W_hh, const float* __restrict__ b_ih,
                const float* __restrict__ b_hh, const float* __restrict__ W_fc,
                const float* __restrict__ b_fc, float* __restrict__ out) {
  // double-buffered hidden state tile: [buf][16 rows][64 cols] f16
  __shared__ _Float16 hbuf[2][16 * HID];

  const int tid  = threadIdx.x;
  const int w    = tid >> 5;      // wave 0..3 -> hidden cols [16w, 16w+16)
  const int lane = tid & 31;
  const int ln   = lane & 15;
  const int hi   = lane >> 4;     // half-wave select
  const int b0   = blockIdx.x * 16;

  // zero-init h0 (both buffers, cheap)
  for (int i = tid; i < 2 * 16 * HID; i += 128)
    (&hbuf[0][0])[i] = (_Float16)0.0f;

  // ---- weight B-fragments pinned in registers (f16), bias folded per lane ----
  // gate gi occupies gate columns [64*gi, 64*gi+64); this wave takes 16 of them.
  v16h  wih[4];        // K = 0..31 (F dimension)
  v16h  whh[4][2];     // K = 0..31 and 32..63 (H dimension)
  float bias[4];
#pragma unroll
  for (int gi = 0; gi < 4; ++gi) {
    const int col = 64 * gi + 16 * w + ln;      // gate column handled by this lane
    wih[gi] = load_cvt_B(W_ih + col * F_IN + hi * 8);
#pragma unroll
    for (int j = 0; j < 2; ++j)
      whh[gi][j] = load_cvt_B(W_hh + col * HID + 32 * j + hi * 8);
    bias[gi] = b_ih[col] + b_hh[col];
  }

  // ---- x A-fragment base: lane ln owns batch row b0+ln; F=32 contiguous ----
  const float* xbase = x + (size_t)(b0 + ln) * T_LEN * F_IN + hi * 8;

  float cst[8];
#pragma unroll
  for (int r = 0; r < 8; ++r) cst[r] = 0.0f;

  v16h xa = load_cvt_A(xbase);          // t = 0, pipelined one step ahead

  __syncthreads();                      // h0 zeros visible

  for (int t = 0; t < T_LEN; ++t) {
    const int rb = t & 1;
    const int wb = rb ^ 1;

    // h A-fragments from LDS (layout matches 16-bit WMMA A striping)
    const _Float16* hrp = &hbuf[rb][ln * HID + hi * 8];
    v16h ha0 = load_A_lds(hrp);         // K = 0..31
    v16h ha1 = load_A_lds(hrp + 32);    // K = 32..63

    // software-pipeline next x fragment (uniform index, no divergence)
    const int tn = (t + 1 < T_LEN) ? (t + 1) : t;
    v16h xn = load_cvt_A(xbase + (size_t)tn * F_IN);

    // 12 WMMAs: gates[16x16] = x@Wih^T + h@Whh^T for i,f,g,o slices
    v8f acc[4];
#pragma unroll
    for (int gi = 0; gi < 4; ++gi) {
      v8f a = {};
      a = __builtin_amdgcn_wmma_f32_16x16x32_f16(false, xa,  false, wih[gi],    (short)0, a, false, false);
      a = __builtin_amdgcn_wmma_f32_16x16x32_f16(false, ha0, false, whh[gi][0], (short)0, a, false, false);
      a = __builtin_amdgcn_wmma_f32_16x16x32_f16(false, ha1, false, whh[gi][1], (short)0, a, false, false);
      acc[gi] = a;
    }

    // pointwise LSTM cell entirely in registers; c never leaves VGPRs
    _Float16* hw = &hbuf[wb][0];
#pragma unroll
    for (int r = 0; r < 8; ++r) {
      const float ig = sigm_fast(acc[0][r] + bias[0]);
      const float fg = sigm_fast(acc[1][r] + bias[1]);
      const float gg = tanh_fast(acc[2][r] + bias[2]);
      const float og = sigm_fast(acc[3][r] + bias[3]);
      const float cn = fg * cst[r] + ig * gg;
      cst[r] = cn;
      const float hv = og * tanh_fast(cn);
      // C/D layout: lane holds col N = ln, row M = r + 8*hi
      hw[(r + 8 * hi) * HID + 16 * w + ln] = (_Float16)hv;
    }

    xa = xn;
    __syncthreads();   // publish h_t before step t+1 reads / overwrites old buf
  }

  // ---- FC head: out[16 x 24] = h_last @ W_fc^T + b_fc (tiny, plain VALU) ----
  // after 512 steps the final h lives in hbuf[0] (512 is even)
  for (int o = tid; o < 16 * HZ; o += 128) {
    const int m = o / HZ;
    const int j = o - m * HZ;
    float s = b_fc[j];
    const _Float16* hp = &hbuf[0][m * HID];
    const float*    wp = W_fc + j * HID;
#pragma unroll
    for (int k = 0; k < HID; ++k) s += (float)hp[k] * wp[k];
    out[(size_t)(b0 + m) * HZ + j] = s;
  }
}

extern "C" void kernel_launch(void* const* d_in, const int* in_sizes, int n_in,
                              void* d_out, int out_size, void* d_ws, size_t ws_size,
                              hipStream_t stream) {
  const float* x    = (const float*)d_in[0];
  const float* W_ih = (const float*)d_in[1];
  const float* W_hh = (const float*)d_in[2];
  const float* b_ih = (const float*)d_in[3];
  const float* b_hh = (const float*)d_in[4];
  const float* W_fc = (const float*)d_in[5];
  const float* b_fc = (const float*)d_in[6];
  float* out = (float*)d_out;

  dim3 grid(B_TOT / 16);   // 256 workgroups, 16 batch rows each
  dim3 block(128);         // 4 wave32s
  hipLaunchKernelGGL(lstm_fused_wmma, grid, block, 0, stream,
                     x, W_ih, W_hh, b_ih, b_hh, W_fc, b_fc, out);
}
